// NodeEdgeBlock_12833362280630
// MI455X (gfx1250) — matrix operand order repaired and checked by reading, not compiled, not run on hardware
//
#include <hip/hip_runtime.h>
#include <math.h>

// ---------------------------------------------------------------------------
// NodeEdgeBlock fused kernel for gfx1250 (MI455X), bf16 WMMA datapath.
//   N=512, XDIM=256, EDIM=128, H=8, DF=32
// Pipeline:
//   1) qkv_kernel  : Q/K/V = x@W.T + b            (f32, tiny)
//   2) pack_kernel : Wem/Wea/Weo -> bf16 B-fragment-major layout (WMMA B layout)
//   3) main_kernel : fused  E1/E2 GEMM (v_wmma_f32_16x16x32_bf16)
//                    -> Y -> newE GEMM (WMMA) streamed to HBM
//                    -> online softmax partials (max/sum/weighted-V) to ws
//   4) final_kernel: merge partials, weighted_V, newX = wV@Wxo.T + bxo
// Round-2 change: half-outer/jt-inner loop order so softmax stats use 48 (not
// 96) persistent VGPRs -> no scratch spills, no VGPR-MSB addressing.
// ---------------------------------------------------------------------------

typedef __bf16 bf16;
typedef bf16  v16bf __attribute__((ext_vector_type(16)));
typedef bf16  v8bf  __attribute__((ext_vector_type(8)));
typedef float v8f   __attribute__((ext_vector_type(8)));

#define NN    512
#define XD    256
#define ED    128
#define JSPLIT 8          // j-splits (grid.y); each covers 512/8 = 64 j = 4 tiles
#define JTILES_PER_WG 4

// workspace byte offsets
#define WS_Q   0u
#define WS_K   524288u
#define WS_V   1048576u
#define WS_PW  1572864u          // packed weights: wem 64KB | wea 64KB | weo 64KB
#define WS_PM  1769472u          // [JSPLIT][512][256] f32 partial max
#define WS_PS  5963776u          // partial sum(exp)
#define WS_PA  10158080u         // partial sum(exp * V)

// LDS byte offsets (dynamic shared)
#define LDS_WEM 0
#define LDS_WEA 65536
#define LDS_WEO 131072
#define LDS_Y   196608           // 8 waves * 16 rows * 264 bf16 (pitch 528B)
#define LDS_TOTAL 264192
#define YPITCH 528               // 264 bf16 elements -> conflict-free b128 reads

// ---------------------------------------------------------------------------
__global__ void qkv_kernel(const float* __restrict__ x,
                           const float* __restrict__ Wq, const float* __restrict__ bq,
                           const float* __restrict__ Wk, const float* __restrict__ bk,
                           const float* __restrict__ Wv, const float* __restrict__ bv,
                           float* __restrict__ Q, float* __restrict__ K,
                           float* __restrict__ V) {
  __shared__ float xs[XD];
  const int i = blockIdx.x, which = blockIdx.y, o = threadIdx.x;
  xs[o] = x[(size_t)i * XD + o];
  __syncthreads();
  const float* W = (which == 0) ? Wq : (which == 1) ? Wk : Wv;
  const float* b = (which == 0) ? bq : (which == 1) ? bk : bv;
  float*       out = (which == 0) ? Q : (which == 1) ? K : V;
  const float* wr = W + (size_t)o * XD;
  float s = b[o];
  #pragma unroll 8
  for (int c = 0; c < XD; ++c) s += xs[c] * wr[c];
  out[(size_t)i * XD + o] = s;
}

// Pack W [rows x cols] (row-major, out-dim major as in torch Linear) into
// WMMA-B fragment-major bf16: frag f = ct*kk32 + kk holds B(32xK,16xN) for
// n = ct*16 + lane%16, k = kk*32 + (lane/16)*8 + t (t<8), +16 + (t-8) (t>=8).
__global__ void pack_kernel(const float* __restrict__ W, int kk32, int ld,
                            bf16* __restrict__ dst) {
  const int f = blockIdx.x, ln = threadIdx.x;
  const int ct = f / kk32, kk = f % kk32;
  const int n = ct * 16 + (ln & 15);
  const int kbase = kk * 32 + (ln >> 4) * 8;
  const float* src = W + (size_t)n * ld + kbase;
  bf16* d = dst + (size_t)f * 512 + ln * 16;
  #pragma unroll
  for (int t = 0; t < 16; ++t) d[t] = (bf16)src[t < 8 ? t : t + 8];
}

// ---------------------------------------------------------------------------
__device__ inline v16bf pack_a16(float4 f0, float4 f1, float4 f2, float4 f3) {
  v16bf a;
  a[0]=(bf16)f0.x;  a[1]=(bf16)f0.y;  a[2]=(bf16)f0.z;  a[3]=(bf16)f0.w;
  a[4]=(bf16)f1.x;  a[5]=(bf16)f1.y;  a[6]=(bf16)f1.z;  a[7]=(bf16)f1.w;
  a[8]=(bf16)f2.x;  a[9]=(bf16)f2.y;  a[10]=(bf16)f2.z; a[11]=(bf16)f2.w;
  a[12]=(bf16)f3.x; a[13]=(bf16)f3.y; a[14]=(bf16)f3.z; a[15]=(bf16)f3.w;
  return a;
}

__device__ inline v16bf lds_b_frag(const char* p) {   // 32B contiguous per lane
  v16bf b;
  ((v8bf*)&b)[0] = ((const v8bf*)p)[0];
  ((v8bf*)&b)[1] = ((const v8bf*)p)[1];
  return b;
}

__global__ __launch_bounds__(256, 1)
void main_kernel(const float* __restrict__ e,
                 const float* __restrict__ Qm, const float* __restrict__ Km,
                 const float* __restrict__ Vm,
                 const float* __restrict__ pw,            // packed bf16 weights (192KB)
                 const float* __restrict__ bem, const float* __restrict__ bea,
                 const float* __restrict__ beo,
                 float* __restrict__ newE,
                 float* __restrict__ pM, float* __restrict__ pS,
                 float* __restrict__ pA) {
  extern __shared__ char lds[];
  const int tid = threadIdx.x;
  const int ln  = tid & 31;          // lane (wave32)
  const int w   = tid >> 5;          // wave 0..7
  const int lc  = ln & 15;           // column within 16x16 C tile
  const int hi  = ln >> 4;           // lane half (C rows +8, A/B K-half select)

  // cooperative copy of packed weights into LDS (192KB = 12288 x 16B)
  {
    const float4* src = (const float4*)pw;
    float4*       dst = (float4*)lds;
    #pragma unroll
    for (int it = 0; it < 48; ++it) dst[tid + it * 256] = src[tid + it * 256];
  }
  __syncthreads();

  const int i0    = blockIdx.x * 16;
  const int jbase = blockIdx.y * 64;
  const float rscale = 0.17677669529663687f;   // 1/sqrt(DF)

  char* ypatch = lds + LDS_Y + w * (16 * YPITCH);   // wave-private Y staging

  #pragma unroll 1
  for (int half = 0; half < 2; ++half) {
    const int m  = half * 8 + w;       // i_local (fixed per M-tile, fixed per half)
    const int iG = i0 + m;
    const float* qrow = Qm + (size_t)iG * XD;

    // online-softmax running stats for this half's i; lane carries c = ct*16+lc
    float smM[16], smS[16], smA[16];
    #pragma unroll
    for (int c = 0; c < 16; ++c) { smM[c] = -1e30f; smS[c] = 0.f; smA[c] = 0.f; }

    #pragma unroll 1
    for (int jt = 0; jt < JTILES_PER_WG; ++jt) {
      const int j0 = jbase + jt * 16;

      // ---- A fragments from e (f32 -> bf16), kk = 0..3 over EDIM=128 ----
      v16bf afr[4];
      {
        const float* erow = e + ((size_t)iG * NN + (size_t)(j0 + lc)) * ED + hi * 8;
        #pragma unroll
        for (int kk = 0; kk < 4; ++kk) {
          const float* p0 = erow + kk * 32;
          float4 f0 = *(const float4*)(p0);
          float4 f1 = *(const float4*)(p0 + 4);
          float4 f2 = *(const float4*)(p0 + 16);
          float4 f3 = *(const float4*)(p0 + 20);
          afr[kk] = pack_a16(f0, f1, f2, f3);
        }
      }

      const float* krow = Km + (size_t)(j0 + hi * 8) * XD + lc;
      const float* vrow = Vm + (size_t)(j0 + hi * 8) * XD + lc;

      // ---- per col-tile: E1/E2 WMMA, Y, online softmax, Y->LDS (bf16) ----
      #pragma unroll
      for (int ct = 0; ct < 16; ++ct) {
        v8f acc1 = {0.f,0.f,0.f,0.f,0.f,0.f,0.f,0.f};
        v8f acc2 = acc1;
        #pragma unroll
        for (int kk = 0; kk < 4; ++kk) {
          v16bf b1 = lds_b_frag(lds + LDS_WEM + (ct * 4 + kk) * 1024 + ln * 32);
          v16bf b2 = lds_b_frag(lds + LDS_WEA + (ct * 4 + kk) * 1024 + ln * 32);
          acc1 = __builtin_amdgcn_wmma_f32_16x16x32_bf16(
                     false, afr[kk], false, b1, (short)0, acc1, false, false);
          acc2 = __builtin_amdgcn_wmma_f32_16x16x32_bf16(
                     false, afr[kk], false, b2, (short)0, acc2, false, false);
        }
        const int c = ct * 16 + lc;
        const float q    = qrow[c];
        const float bemc = bem[c];
        const float beac = bea[c];

        float y[8], vv[8];
        float tmax = -1e30f;
        #pragma unroll
        for (int r = 0; r < 8; ++r) {
          const float kv = krow[(size_t)r * XD + ct * 16];
          vv[r] = vrow[(size_t)r * XD + ct * 16];
          const float yy = q * kv * rscale * (acc1[r] + bemc + 1.0f) + (acc2[r] + beac);
          y[r] = yy;
          tmax = fmaxf(tmax, yy);
        }
        tmax = fmaxf(tmax, __shfl_xor(tmax, 16, 32));
        const float mo = smM[ct];
        const float mn = fmaxf(mo, tmax);
        const float fs = __expf(mo - mn);
        float dsum = 0.f, dacc = 0.f;
        #pragma unroll
        for (int r = 0; r < 8; ++r) {
          const float p = __expf(y[r] - mn);
          dsum += p;
          dacc += p * vv[r];
          *(bf16*)(ypatch + (size_t)(r + hi * 8) * YPITCH + c * 2) = (bf16)y[r];
        }
        dsum += __shfl_xor(dsum, 16, 32);
        dacc += __shfl_xor(dacc, 16, 32);
        smM[ct] = mn;
        smS[ct] = smS[ct] * fs + dsum;
        smA[ct] = smA[ct] * fs + dacc;
      }

      // ---- newE = Y @ Weo.T + beo  (wave-private: DS ops in-order per wave) ----
      v8f ne[8];
      #pragma unroll
      for (int nt = 0; nt < 8; ++nt) ne[nt] = (v8f){0.f,0.f,0.f,0.f,0.f,0.f,0.f,0.f};
      #pragma unroll
      for (int kk = 0; kk < 8; ++kk) {
        const char* ap = ypatch + (size_t)lc * YPITCH + (kk * 32 + hi * 8) * 2;
        v16bf a;
        ((v8bf*)&a)[0] = *(const v8bf*)(ap);
        ((v8bf*)&a)[1] = *(const v8bf*)(ap + 32);   // +16 elements (k+16)
        #pragma unroll
        for (int nt = 0; nt < 8; ++nt) {
          v16bf b = lds_b_frag(lds + LDS_WEO + (nt * 8 + kk) * 1024 + ln * 32);
          ne[nt] = __builtin_amdgcn_wmma_f32_16x16x32_bf16(
                       false, a, false, b, (short)0, ne[nt], false, false);
        }
      }
      {
        float* erowout = newE + ((size_t)iG * NN + (size_t)(j0 + hi * 8)) * ED + lc;
        #pragma unroll
        for (int nt = 0; nt < 8; ++nt) {
          const float bc = beo[nt * 16 + lc];
          #pragma unroll
          for (int r = 0; r < 8; ++r) {
            erowout[(size_t)r * ED + nt * 16] = ne[nt][r] + bc;
          }
        }
      }
    }

    // ---- write this half's softmax partials (lanes 0..15 carry c) ----
    if (hi == 0) {
      const size_t base = ((size_t)blockIdx.y * NN + iG) * XD;
      #pragma unroll
      for (int ct = 0; ct < 16; ++ct) {
        pM[base + ct * 16 + lc] = smM[ct];
        pS[base + ct * 16 + lc] = smS[ct];
        pA[base + ct * 16 + lc] = smA[ct];
      }
    }
  }
}

// ---------------------------------------------------------------------------
__global__ void final_kernel(const float* __restrict__ pM, const float* __restrict__ pS,
                             const float* __restrict__ pA,
                             const float* __restrict__ Wxo, const float* __restrict__ bxo,
                             float* __restrict__ newX) {
  __shared__ float wv[XD];
  const int i = blockIdx.x, t = threadIdx.x;
  float m = -1e30f;
  #pragma unroll
  for (int s = 0; s < JSPLIT; ++s)
    m = fmaxf(m, pM[((size_t)s * NN + i) * XD + t]);
  float ssum = 0.f, asum = 0.f;
  #pragma unroll
  for (int s = 0; s < JSPLIT; ++s) {
    const size_t idx = ((size_t)s * NN + i) * XD + t;
    const float f = __expf(pM[idx] - m);
    ssum += pS[idx] * f;
    asum += pA[idx] * f;
  }
  wv[t] = asum / ssum;
  __syncthreads();
  const float* wr = Wxo + (size_t)t * XD;
  float acc = bxo[t];
  #pragma unroll 8
  for (int c = 0; c < XD; ++c) acc += wv[c] * wr[c];
  newX[(size_t)i * XD + t] = acc;
}

// ---------------------------------------------------------------------------
extern "C" void kernel_launch(void* const* d_in, const int* in_sizes, int n_in,
                              void* d_out, int out_size, void* d_ws, size_t ws_size,
                              hipStream_t stream) {
  const float* x   = (const float*)d_in[0];
  const float* e   = (const float*)d_in[1];
  // d_in[2] = adj (unused by reference)
  const float* Wq  = (const float*)d_in[3];
  const float* bq  = (const float*)d_in[4];
  const float* Wk  = (const float*)d_in[5];
  const float* bk  = (const float*)d_in[6];
  const float* Wv  = (const float*)d_in[7];
  const float* bv  = (const float*)d_in[8];
  const float* Wem = (const float*)d_in[9];
  const float* bem = (const float*)d_in[10];
  const float* Wea = (const float*)d_in[11];
  const float* bea = (const float*)d_in[12];
  const float* Wxo = (const float*)d_in[13];
  const float* bxo = (const float*)d_in[14];
  const float* Weo = (const float*)d_in[15];
  const float* beo = (const float*)d_in[16];

  float* newX = (float*)d_out;                    // [512,256]
  float* newE = (float*)d_out + (size_t)NN * XD;  // [512,512,128]

  char*  ws = (char*)d_ws;
  float* Q  = (float*)(ws + WS_Q);
  float* K  = (float*)(ws + WS_K);
  float* V  = (float*)(ws + WS_V);
  bf16*  pwem = (bf16*)(ws + WS_PW);
  bf16*  pwea = (bf16*)(ws + WS_PW + 65536);
  bf16*  pweo = (bf16*)(ws + WS_PW + 131072);
  float* PM = (float*)(ws + WS_PM);
  float* PS = (float*)(ws + WS_PS);
  float* PA = (float*)(ws + WS_PA);

  (void)hipFuncSetAttribute((const void*)main_kernel,
                            hipFuncAttributeMaxDynamicSharedMemorySize, LDS_TOTAL);

  qkv_kernel<<<dim3(NN, 3), XD, 0, stream>>>(x, Wq, bq, Wk, bk, Wv, bv, Q, K, V);
  pack_kernel<<<64, 32, 0, stream>>>(Wem, 4, ED, pwem);   // 16 ct x 4 kk
  pack_kernel<<<64, 32, 0, stream>>>(Wea, 4, ED, pwea);   // 16 ct x 4 kk
  pack_kernel<<<64, 32, 0, stream>>>(Weo, 8, XD, pweo);   // 8 nt x 8 kk
  main_kernel<<<dim3(NN / 16, JSPLIT), 256, LDS_TOTAL, stream>>>(
      e, Q, K, V, (const float*)pwem, bem, bea, beo, newE, PM, PS, PA);
  final_kernel<<<NN, XD, 0, stream>>>(PM, PS, PA, Wxo, bxo, newX);
}
